// MultiScaleRetentionCompat_55009941127826
// MI455X (gfx1250) — matrix operand order repaired and checked by reading, compiled, not run on hardware
//
#include <hip/hip_runtime.h>
#include <hip/hip_bf16.h>
#include <math.h>

typedef __bf16 bf16;
typedef __attribute__((ext_vector_type(16))) __bf16 bf16x16;
typedef __attribute__((ext_vector_type(8)))  float   floatx8;
typedef __attribute__((ext_vector_type(4)))  int     i32x4;

#define GLOBAL_AS __attribute__((address_space(1)))
#define SHARED_AS __attribute__((address_space(3)))

#define D_MODEL 2048
#define NTOK    16384   // 4 * 4096 tokens
#define SEQ     4096
#define BATCH   4
#define NHEAD   16
#define DH      128
#define CHUNK   256
#define NCHUNK  16      // SEQ / CHUNK

#define KT      32      // K-tile
#define BM      128     // block tile M
#define BN      128     // block tile N
#define LDS_STRIDE 40   // 32 bf16 + 8 pad

// ---------------- async global -> LDS copy of 16 bytes (8 bf16) ----------------
__device__ __forceinline__ void cp16(bf16* dst, const bf16* src) {
#if __has_builtin(__builtin_amdgcn_global_load_async_to_lds_b128)
  __builtin_amdgcn_global_load_async_to_lds_b128(
      (GLOBAL_AS i32x4*)src,
      (SHARED_AS i32x4*)dst, 0, 0);
#else
  *(float4*)dst = *(const float4*)src;
#endif
}

__device__ __forceinline__ void wait_cp() {
#if __has_builtin(__builtin_amdgcn_global_load_async_to_lds_b128)
#if __has_builtin(__builtin_amdgcn_s_wait_asynccnt)
  __builtin_amdgcn_s_wait_asynccnt(0);
#else
  asm volatile("s_wait_asynccnt 0" ::: "memory");
#endif
#endif
}

// ---------------- fp32 -> bf16 elementwise convert (x) ----------------
__global__ void k_cvt_bf16(const float* __restrict__ in, bf16* __restrict__ out, int n4) {
  int i = blockIdx.x * blockDim.x + threadIdx.x;
  if (i < n4) {
    float4 f = ((const float4*)in)[i];
    bf16* o = out + (size_t)i * 4;
    o[0] = (bf16)f.x; o[1] = (bf16)f.y; o[2] = (bf16)f.z; o[3] = (bf16)f.w;
  }
}

// ---------------- fp32 [K][N] -> bf16 [N][K] transpose-convert ----------------
__global__ void k_transpose_cvt(const float* __restrict__ W, bf16* __restrict__ Wt) {
  __shared__ float tile[32][33];
  int tx = threadIdx.x, ty = threadIdx.y;
  int x = blockIdx.x * 32 + tx;
  int ybase = blockIdx.y * 32;
  #pragma unroll
  for (int j = 0; j < 32; j += 8)
    tile[ty + j][tx] = W[(size_t)(ybase + ty + j) * D_MODEL + x];
  __syncthreads();
  int xo = blockIdx.x * 32;
  #pragma unroll
  for (int j = 0; j < 32; j += 8)
    Wt[(size_t)(xo + ty + j) * D_MODEL + ybase + tx] = (bf16)tile[tx][ty + j];
}

// ---------------- bf16 WMMA GEMM: C[M,2048] = A[M,2048] * Bt[2048,2048]^T ----------------
// Bt is N-major (row n holds K contiguous). Block tile 128x128, K-tile 32,
// 8 waves in 2x4 grid, each wave computes 64x32 (4x2 WMMA accumulators,
// ~140 VGPRs -> no VGPR-MSB mode, good occupancy).
// Double-buffered LDS filled via GLOBAL_LOAD_ASYNC_TO_LDS_B128 (ASYNCcnt).
__global__ __launch_bounds__(256)
void k_gemm_bf16(const bf16* __restrict__ A, const bf16* __restrict__ Bt,
                 float* __restrict__ C) {
  __shared__ __align__(16) bf16 As[2][BM * LDS_STRIDE];
  __shared__ __align__(16) bf16 Bs[2][BN * LDS_STRIDE];

  const int tid  = threadIdx.x;
  const int lane = tid & 31;
  const int wave = tid >> 5;
  const int wrow = wave >> 2, wcol = wave & 3;   // 2x4 wave grid
  const int mBase = wrow * 64, nBase = wcol * 32;
  const int lrow = lane & 15;
  const int hiHalf = lane >> 4;                  // lane half selects K sub-range
  const int aOff = hiHalf ? 16 : 0;              // A frag: K 0-15 vs 16-31
  const int bOff = hiHalf ? 16 : 0;              // B frag: K 0-15 vs 16-31
  const int m0 = blockIdx.y * BM;
  const int n0 = blockIdx.x * BN;

  const int trow = tid >> 1;                     // 0..127 (tile row to load)
  const int tk   = (tid & 1) * 16;               // 0 or 16

  floatx8 acc[4][2];
  const floatx8 zero = {0.f, 0.f, 0.f, 0.f, 0.f, 0.f, 0.f, 0.f};
  #pragma unroll
  for (int i = 0; i < 4; i++)
    #pragma unroll
    for (int j = 0; j < 2; j++) acc[i][j] = zero;

  // tile loader: A rows m0..m0+127 and B rows n0..n0+127,
  // each thread moves 16 bf16 of A and 16 bf16 of B (2+2 async b128)
  auto load_tile = [&](int buf, int k0) {
    const bf16* ga = A + (size_t)(m0 + trow) * D_MODEL + k0 + tk;
    bf16* sa = &As[buf][trow * LDS_STRIDE + tk];
    cp16(sa,     ga);
    cp16(sa + 8, ga + 8);
    const bf16* gb = Bt + (size_t)(n0 + trow) * D_MODEL + k0 + tk;
    bf16* sb = &Bs[buf][trow * LDS_STRIDE + tk];
    cp16(sb,     gb);
    cp16(sb + 8, gb + 8);
  };

  load_tile(0, 0);
  wait_cp();
  __syncthreads();

  for (int kt = 0; kt < D_MODEL / KT; kt++) {
    const int cur = kt & 1;
    if (kt + 1 < D_MODEL / KT) load_tile(cur ^ 1, (kt + 1) * KT);

    const bf16* as = As[cur];
    const bf16* bs = Bs[cur];
    union { bf16x16 v; float4 q[2]; } a[4], b[2];
    #pragma unroll
    for (int nt = 0; nt < 2; nt++) {
      const float4* p = (const float4*)(bs + (nBase + nt * 16 + lrow) * LDS_STRIDE + bOff);
      b[nt].q[0] = p[0]; b[nt].q[1] = p[1];
    }
    #pragma unroll
    for (int mt = 0; mt < 4; mt++) {
      int base = (mBase + mt * 16 + lrow) * LDS_STRIDE + aOff;
      a[mt].q[0] = *(const float4*)(as + base);
      a[mt].q[1] = *(const float4*)(as + base + 8);
    }
    #pragma unroll
    for (int mt = 0; mt < 4; mt++)
      #pragma unroll
      for (int nt = 0; nt < 2; nt++)
        acc[mt][nt] = __builtin_amdgcn_wmma_f32_16x16x32_bf16(
            false, a[mt].v, false, b[nt].v, (short)0, acc[mt][nt], false, false);

    wait_cp();
    __syncthreads();
  }

  #pragma unroll
  for (int mt = 0; mt < 4; mt++) {
    #pragma unroll
    for (int nt = 0; nt < 2; nt++) {
      int row0 = m0 + mBase + mt * 16 + hiHalf * 8;
      int col  = n0 + nBase + nt * 16 + lrow;
      #pragma unroll
      for (int r = 0; r < 8; r++)
        C[(size_t)(row0 + r) * D_MODEL + col] = acc[mt][nt][r];
    }
  }
}

// ---------------- chunked retention scan, pass 1: local scan in-place over v ----------------
__global__ void k_scan_local(float* __restrict__ v, const float* __restrict__ beta,
                             float* __restrict__ finals) {
  int tid = blockIdx.x * blockDim.x + threadIdx.x;   // 131072 total
  int dh = tid & 127;
  int h  = (tid >> 7) & 15;
  int c  = (tid >> 11) & 15;
  int b  = tid >> 15;
  float lam = 1.f / (1.f + __expf(-beta[h]));
  size_t base = ((size_t)b * SEQ + (size_t)c * CHUNK) * D_MODEL + (size_t)h * DH + dh;
  float s = 0.f;
  for (int i = 0; i < CHUNK; i++) {
    float val = v[base];
    s = lam * s + val;
    v[base] = s;                    // states (without carry) in-place
    base += D_MODEL;
  }
  finals[tid] = s;                  // layout matches [b][c][h][dh]
}

// ---------------- pass 2: cross-chunk carry scan; finals[] becomes carry-in ----------------
__global__ void k_scan_carry(float* __restrict__ finals, const float* __restrict__ beta) {
  int tid = blockIdx.x * blockDim.x + threadIdx.x;   // 8192 = (b,h,dh)
  int dh = tid & 127;
  int h  = (tid >> 7) & 15;
  int b  = tid >> 11;
  float lam  = 1.f / (1.f + __expf(-beta[h]));
  float lamL = __powf(lam, (float)CHUNK);
  float G = 0.f;
  for (int c = 0; c < NCHUNK; c++) {
    size_t idx = (((size_t)b * NCHUNK + c) * NHEAD + h) * DH + dh;
    float f = finals[idx];
    finals[idx] = G;                // carry-in for chunk c
    G = f + lamL * G;
  }
}

// ---------------- pass 3 fused: carry fixup + y=q*state + LayerNorm + SiLU gate -> bf16 ----------------
__global__ __launch_bounds__(256)
void k_finalize(const float* __restrict__ q, const float* __restrict__ st,
                const float* __restrict__ carry, const float* __restrict__ gbuf,
                const float* __restrict__ beta, const float* __restrict__ ln_g,
                const float* __restrict__ ln_b, bf16* __restrict__ yb) {
  int tok = blockIdx.x;             // B*S tokens
  int b = tok >> 12;
  int s = tok & 4095;
  int c = s >> 8;
  int i = s & 255;
  int tid = threadIdx.x;
  int h = tid >> 4;                 // 8 elems/thread within one head block
  float lam  = 1.f / (1.f + __expf(-beta[h]));
  float lamp = __powf(lam, (float)(i + 1));
  size_t base  = (size_t)tok * D_MODEL + (size_t)tid * 8;
  size_t cbase = (((size_t)b * NCHUNK + c) * NHEAD + h) * DH + ((tid * 8) & 127);

  float y[8];
  float sum = 0.f, sq = 0.f;
  #pragma unroll
  for (int e = 0; e < 8; e++) {
    float stv = st[base + e] + lamp * carry[cbase + e];
    float vy = q[base + e] * stv;
    y[e] = vy; sum += vy; sq += vy * vy;
  }
  __shared__ float s1[256], s2[256];
  s1[tid] = sum; s2[tid] = sq;
  __syncthreads();
  for (int off = 128; off > 0; off >>= 1) {
    if (tid < off) { s1[tid] += s1[tid + off]; s2[tid] += s2[tid + off]; }
    __syncthreads();
  }
  float mean = s1[0] * (1.f / D_MODEL);
  float var  = s2[0] * (1.f / D_MODEL) - mean * mean;
  float rstd = rsqrtf(var + 1e-5f);
  int d0 = tid * 8;
  #pragma unroll
  for (int e = 0; e < 8; e++) {
    float yn = (y[e] - mean) * rstd * ln_g[d0 + e] + ln_b[d0 + e];
    float g  = gbuf[base + e];
    float sg = g / (1.f + __expf(-g));
    yb[base + e] = (bf16)(yn * sg);
  }
}

extern "C" void kernel_launch(void* const* d_in, const int* in_sizes, int n_in,
                              void* d_out, int out_size, void* d_ws, size_t ws_size,
                              hipStream_t stream) {
  (void)in_sizes; (void)n_in; (void)out_size; (void)ws_size;
  const float* x    = (const float*)d_in[0];
  const float* Wq   = (const float*)d_in[1];
  const float* Wv   = (const float*)d_in[2];
  const float* Wg   = (const float*)d_in[3];
  const float* Wo   = (const float*)d_in[4];
  const float* beta = (const float*)d_in[5];
  const float* ln_g = (const float*)d_in[6];
  const float* ln_b = (const float*)d_in[7];

  char* ws = (char*)d_ws;
  size_t off = 0;
  bf16* xb  = (bf16*)(ws + off); off += (size_t)NTOK * D_MODEL * 2;      // 64 MB
  bf16* Wqt = (bf16*)(ws + off); off += (size_t)D_MODEL * D_MODEL * 2;   // 8 MB
  bf16* Wvt = (bf16*)(ws + off); off += (size_t)D_MODEL * D_MODEL * 2;
  bf16* Wgt = (bf16*)(ws + off); off += (size_t)D_MODEL * D_MODEL * 2;
  bf16* Wot = (bf16*)(ws + off); off += (size_t)D_MODEL * D_MODEL * 2;
  float* qbuf   = (float*)(ws + off); off += (size_t)NTOK * D_MODEL * 4; // 128 MB
  float* vbuf   = (float*)(ws + off); off += (size_t)NTOK * D_MODEL * 4; // states in-place
  float* gbuf   = (float*)(ws + off); off += (size_t)NTOK * D_MODEL * 4;
  float* finals = (float*)(ws + off); off += (size_t)BATCH * NCHUNK * NHEAD * DH * 4;
  bf16* yb  = (bf16*)(ws + off); off += (size_t)NTOK * D_MODEL * 2;

  // precision conversion / weight transpose
  k_cvt_bf16<<<(NTOK * D_MODEL / 4 + 255) / 256, 256, 0, stream>>>(x, xb, NTOK * D_MODEL / 4);
  dim3 tb(32, 8), tg(64, 64);
  k_transpose_cvt<<<tg, tb, 0, stream>>>(Wq, Wqt);
  k_transpose_cvt<<<tg, tb, 0, stream>>>(Wv, Wvt);
  k_transpose_cvt<<<tg, tb, 0, stream>>>(Wg, Wgt);
  k_transpose_cvt<<<tg, tb, 0, stream>>>(Wo, Wot);

  // three input GEMMs
  dim3 gg(D_MODEL / BN, NTOK / BM);   // (16, 128)
  k_gemm_bf16<<<gg, 256, 0, stream>>>(xb, Wqt, qbuf);
  k_gemm_bf16<<<gg, 256, 0, stream>>>(xb, Wvt, vbuf);
  k_gemm_bf16<<<gg, 256, 0, stream>>>(xb, Wgt, gbuf);

  // retention scan (chunked)
  k_scan_local<<<(BATCH * NCHUNK * NHEAD * DH + 255) / 256, 256, 0, stream>>>(vbuf, beta, finals);
  k_scan_carry<<<(BATCH * NHEAD * DH + 255) / 256, 256, 0, stream>>>(finals, beta);

  // fused carry fixup + elementwise + LN + gate -> bf16
  k_finalize<<<NTOK, 256, 0, stream>>>(qbuf, vbuf, finals, gbuf, beta, ln_g, ln_b, yb);

  // output projection
  k_gemm_bf16<<<gg, 256, 0, stream>>>(yb, Wot, (float*)d_out);
}
